// PVPNet_11776800325757
// MI455X (gfx1250) — matrix operand order repaired and checked by reading, not compiled, not run on hardware
//
#include <hip/hip_runtime.h>
#include <hip/hip_bf16.h>

typedef _Float16 v16h __attribute__((ext_vector_type(16)));
typedef _Float16 v8h  __attribute__((ext_vector_type(8)));
typedef float    v8f  __attribute__((ext_vector_type(8)));
typedef unsigned int v4u __attribute__((ext_vector_type(4)));
typedef int      v8i __attribute__((ext_vector_type(8)));
typedef int      v4i __attribute__((ext_vector_type(4)));

#define BB 32
#define NP 2048
#define WGRID 32
#define GV (WGRID*WGRID*WGRID)
#define BNEPS 1e-5f

#if defined(__has_builtin)
#  if __has_builtin(__builtin_amdgcn_tensor_load_to_lds) && \
      __has_builtin(__builtin_amdgcn_s_wait_tensorcnt)
#    define USE_TDM 1
#  endif
#endif
#ifndef USE_TDM
#  define USE_TDM 0
#endif

static __device__ __forceinline__ v16h cat8(v8h lo, v8h hi) {
  return __builtin_shufflevector(lo, hi, 0,1,2,3,4,5,6,7,8,9,10,11,12,13,14,15);
}

// ---------------------------------------------------------------------------
// Weight packing: wpack[((cot*nk + kc)*32 + lane)*16 + e] = W in the exact
// per-lane B-fragment layout this file feeds v_wmma_f32_16x16x32_f16 (contiguous k, as A).
//   K = kc*32 + 16*(lane/16) + e ; co = cot*16 + (lane%16)
// Conv weights (co, ci, 4,4,4): K = tap*Cin + ci, tap=(kd*4+kh)*4+kw.
// ---------------------------------------------------------------------------
__global__ void k_pack_conv(const float* __restrict__ w, _Float16* __restrict__ dst,
                            int Cin, int Co) {
  size_t tid = (size_t)blockIdx.x * blockDim.x + threadIdx.x;
  size_t total = (size_t)Co * Cin * 64;
  if (tid >= total) return;
  int nk = (Cin * 64) >> 5;
  int e    = (int)(tid & 15);
  int lane = (int)((tid >> 4) & 31);
  int kc   = (int)((tid >> 9) % nk);
  int cot  = (int)(tid / ((size_t)512 * nk));
  int K  = kc * 32 + ((lane >> 4) << 4) + e;
  int co = cot * 16 + (lane & 15);
  int tap = K / Cin;
  int ci  = K - tap * Cin;
  dst[tid] = (_Float16)w[((size_t)co * Cin + ci) * 64 + tap];
}

__global__ void k_pack_fc(const float* __restrict__ w, _Float16* __restrict__ dst,
                          int Ktot, int Co) {
  size_t tid = (size_t)blockIdx.x * blockDim.x + threadIdx.x;
  size_t total = (size_t)Co * Ktot;
  if (tid >= total) return;
  int nk = Ktot >> 5;
  int e    = (int)(tid & 15);
  int lane = (int)((tid >> 4) & 31);
  int kc   = (int)((tid >> 9) % nk);
  int cot  = (int)(tid / ((size_t)512 * nk));
  int K  = kc * 32 + ((lane >> 4) << 4) + e;
  int co = cot * 16 + (lane & 15);
  dst[tid] = (_Float16)w[(size_t)co * Ktot + K];
}

// Fold conv-bias + BatchNorm into per-channel scale/shift.
__global__ void k_bnprep(const float* __restrict__ g, const float* __restrict__ be,
                         const float* __restrict__ m, const float* __restrict__ v,
                         const float* __restrict__ cb,
                         float* __restrict__ scale, float* __restrict__ shift, int C) {
  int c = blockIdx.x * blockDim.x + threadIdx.x;
  if (c >= C) return;
  float s = g[c] * rsqrtf(v[c] + BNEPS);
  scale[c] = s;
  shift[c] = (cb[c] - m[c]) * s + be[c];
}

// Point MLP (3->64 BN ReLU ->64) + scatter-add into voxel sums/counts.
__global__ void k_point_scatter(const float* __restrict__ x,
    const float* __restrict__ w1, const float* __restrict__ b1,
    const float* __restrict__ g0, const float* __restrict__ be0,
    const float* __restrict__ m0, const float* __restrict__ v0,
    const float* __restrict__ w2, const float* __restrict__ b2,
    float* __restrict__ sums, float* __restrict__ cnt) {
  int tid = blockIdx.x * blockDim.x + threadIdx.x;
  if (tid >= BB * NP) return;
  const float* p = x + (size_t)tid * 3;
  float px = p[0], py = p[1], pz = p[2];
  const float d = 2.0f / WGRID;
  int u  = min(max((int)floorf((px + 1.0f) / d), 0), WGRID - 1);
  int kk = min(max((int)floorf((py + 1.0f) / d), 0), WGRID - 1);
  int vv = min(max((int)floorf((1.0f - pz) / d), 0), WGRID - 1);
  int idx = u * WGRID * WGRID + vv * WGRID + kk;
  float h1[64];
  #pragma unroll 4
  for (int c = 0; c < 64; ++c) {
    float h = px * w1[c * 3 + 0] + py * w1[c * 3 + 1] + pz * w1[c * 3 + 2] + b1[c];
    float s = g0[c] * rsqrtf(v0[c] + BNEPS);
    h = (h - m0[c]) * s + be0[c];
    h1[c] = fmaxf(h, 0.0f);
  }
  int b = tid / NP;
  float* dstp = sums + ((size_t)b * GV + idx) * 64;
  for (int c2 = 0; c2 < 64; ++c2) {
    float f = b2[c2];
    const float* wr = w2 + (size_t)c2 * 64;
    #pragma unroll 8
    for (int c = 0; c < 64; ++c) f += h1[c] * wr[c];
    atomicAdd(dstp + c2, f);
  }
  atomicAdd(cnt + (size_t)b * GV + idx, 1.0f);
}

// sums/max(cnt,1) -> f16 channel-last voxel grid.
__global__ void k_normalize(const float* __restrict__ sums, const float* __restrict__ cnt,
                            _Float16* __restrict__ out) {
  size_t tid = (size_t)blockIdx.x * blockDim.x + threadIdx.x;
  if (tid >= (size_t)BB * GV * 64) return;
  float c = cnt[tid >> 6];
  out[tid] = (_Float16)(sums[tid] / fmaxf(c, 1.0f));
}

// ---------------------------------------------------------------------------
// Implicit-GEMM Conv3d(k=4,pad=2) + folded BN + LeakyReLU(0.2) via WMMA.
// Activations channel-last f16 [b][z][y][x][c].
// Block = 4 waves sharing one co-pair. Per 8-chunk K-block, 16KB of packed
// B-fragments are staged into LDS — via the Tensor Data Mover (2-row 2D tile,
// row = 8KB, row-stride = nk*512 elements) when available, else cooperative
// b128 copies. Each wave runs 16 positions x 32 output channels.
// LDS layout: smem[tsel][KCB*512] (rows concatenated, matching TDM fill order).
// ---------------------------------------------------------------------------
#define KCB 8
__global__ __launch_bounds__(128)
void k_conv_wmma(const _Float16* __restrict__ in, const _Float16* __restrict__ wp,
                 const float* __restrict__ scale, const float* __restrict__ shift,
                 _Float16* __restrict__ out, int Cin, int Co, int ID, int OD) {
  __shared__ __align__(16) _Float16 smem[2 * KCB * 512];   // 16 KB
  const int lane = threadIdx.x & 31;
  const int wid  = threadIdx.x >> 5;
  const int P = OD * OD * OD;
  const int pt = blockIdx.x * 4 + wid;            // wave-uniform
  const bool active = pt * 16 < P;                // keep all waves for barriers
  const int cp  = blockIdx.y;                     // co-pair
  const int b   = blockIdx.z;
  const int row = lane & 15;
  const int p   = pt * 16 + row;
  const bool rv = active && (p < P);
  const int pc  = (p < P) ? p : 0;
  const int od = pc / (OD * OD);
  const int r2 = pc - od * OD * OD;
  const int oh = r2 / OD, ow = r2 - (r2 / OD) * OD;
  const int kbase = (lane >> 4) << 4;             // 0 or 16
  const int cpt = Cin >> 5;                       // K-chunks per tap
  const int nk  = Cin << 1;                       // Cin*64/32
  v8f acc0 = {}, acc1 = {};
  for (int kb = 0; kb < nk; kb += KCB) {
#if USE_TDM
    if (wid == 0) {
      // 2D TDM descriptor: 2 rows x 4096 f16, row stride nk*512 elements.
      unsigned long long ga =
          (unsigned long long)(const void*)(wp + (size_t)(cp * 2 * nk + kb) * 512);
      unsigned int ldsb = (unsigned int)(size_t)(&smem[0]);   // LDS aperture low 32
      unsigned long long strd = (unsigned long long)nk * 512; // elements
      v4u g0;
      g0.x = 1u;                                   // count=1 valid descriptor
      g0.y = ldsb;                                 // lds_addr
      g0.z = (unsigned int)ga;                     // global_addr[31:0]
      g0.w = (unsigned int)((ga >> 32) & 0x1FFFFFFu) | (2u << 30); // [56:32]|type=2
      v8i g1;
      g1[0] = (int)(1u << 16);                     // data_size = 2 bytes
      g1[1] = (int)((4096u & 0xFFFFu) << 16);      // tensor_dim0 @bit48
      g1[2] = (int)(((4096u >> 16) & 0xFFFFu) | (2u << 16)); // dim0 hi | tensor_dim1 lo
      g1[3] = (int)(4096u << 16);                  // tensor_dim1 hi=0 | tile_dim0 @112
      g1[4] = 2;                                   // tile_dim1=2, tile_dim2=0
      g1[5] = (int)(unsigned int)(strd & 0xFFFFFFFFull);      // dim0_stride low32
      g1[6] = (int)(unsigned int)((strd >> 32) & 0xFFFFull);  // dim0_stride hi16
      g1[7] = 0;
      v4i gz = {0, 0, 0, 0};
      v8i gz8 = {0, 0, 0, 0, 0, 0, 0, 0};
      __builtin_amdgcn_tensor_load_to_lds(g0, g1, gz, gz, gz8, 0);
      __builtin_amdgcn_s_wait_tensorcnt(0);
    }
#else
    // Cooperative stage: 1024 x 16B vectors, layout [tsel][j][512].
    #pragma unroll
    for (int i = 0; i < 8; ++i) {
      int vec  = i * 128 + threadIdx.x;
      int tsel = vec >> 9;
      int rem  = vec & 511;
      int j    = rem >> 6;
      int v    = rem & 63;
      int cot  = cp * 2 + tsel;
      const _Float16* src = wp + (size_t)(cot * nk + kb + j) * 512 + v * 8;
      _Float16* dst = smem + (tsel * KCB + j) * 512 + v * 8;
      *(v8h*)dst = *(const v8h*)src;
    }
#endif
    __syncthreads();
    if (active) {
      #pragma unroll
      for (int j = 0; j < KCB; ++j) {
        int kc = kb + j;
        int tap = kc / cpt;
        int cb  = (kc - tap * cpt) << 5;
        int kd = tap >> 4, kh = (tap >> 2) & 3, kw = tap & 3;
        int z = od - 2 + kd, y = oh - 2 + kh, xx = ow - 2 + kw;
        v16h a = {};
        if (rv && (unsigned)z < (unsigned)ID && (unsigned)y < (unsigned)ID &&
            (unsigned)xx < (unsigned)ID) {
          const _Float16* s = in + ((((size_t)b * ID + z) * ID + y) * ID + xx) * Cin + cb + kbase;
          a = cat8(*(const v8h*)s, *(const v8h*)(s + 8));
        }
        const _Float16* w0 = smem + j * 512 + lane * 16;
        const _Float16* w1 = smem + (KCB + j) * 512 + lane * 16;
        v16h bf0 = cat8(*(const v8h*)w0, *(const v8h*)(w0 + 8));
        v16h bf1 = cat8(*(const v8h*)w1, *(const v8h*)(w1 + 8));
        acc0 = __builtin_amdgcn_wmma_f32_16x16x32_f16(false, a, false, bf0,
                                                      (short)0, acc0, false, false);
        acc1 = __builtin_amdgcn_wmma_f32_16x16x32_f16(false, a, false, bf1,
                                                      (short)0, acc1, false, false);
      }
    }
    __syncthreads();
  }
  if (!active) return;
  const int co0 = (cp * 2) * 16 + (lane & 15);
  const int co1 = co0 + 16;
  const float sc0 = scale[co0], sh0 = shift[co0];
  const float sc1 = scale[co1], sh1 = shift[co1];
  const int mo = (lane >> 4) << 3;
  #pragma unroll
  for (int r = 0; r < 8; ++r) {
    int pp = pt * 16 + r + mo;
    if (pp < P) {
      float v0 = acc0[r] * sc0 + sh0;
      v0 = v0 > 0.0f ? v0 : 0.2f * v0;
      float v1 = acc1[r] * sc1 + sh1;
      v1 = v1 > 0.0f ? v1 : 0.2f * v1;
      size_t base = ((size_t)b * P + pp) * Co;
      out[base + co0] = (_Float16)v0;
      out[base + co1] = (_Float16)v1;
    }
  }
}

__global__ void k_maxpool(const _Float16* __restrict__ in, _Float16* __restrict__ out,
                          int C, int OD, int PD) {
  size_t tid = (size_t)blockIdx.x * blockDim.x + threadIdx.x;
  size_t total = (size_t)BB * PD * PD * PD * C;
  if (tid >= total) return;
  int c = (int)(tid % C); size_t t = tid / C;
  int pw = (int)(t % PD); t /= PD;
  int ph = (int)(t % PD); t /= PD;
  int pd = (int)(t % PD);
  int b  = (int)(t / PD);
  float mx = -3.0e38f;
  for (int dz = 0; dz < 2; ++dz)
    for (int dy = 0; dy < 2; ++dy)
      for (int dx = 0; dx < 2; ++dx) {
        size_t o = ((((size_t)b * OD + pd * 2 + dz) * OD + ph * 2 + dy) * OD + pw * 2 + dx) * C + c;
        mx = fmaxf(mx, (float)in[o]);
      }
  out[tid] = (_Float16)mx;
}

// channel-last (B,2,2,2,512) -> NCDHW-flatten order K = c*8 + (d*4+h*2+w)
__global__ void k_flatten(const _Float16* __restrict__ in, _Float16* __restrict__ out) {
  int tid = blockIdx.x * blockDim.x + threadIdx.x;
  if (tid >= BB * 4096) return;
  int b = tid >> 12, K = tid & 4095;
  int c = K >> 3, s = K & 7;
  out[tid] = in[((size_t)b * 8 + s) * 512 + c];
}

// fc1: (32x4096)x(4096x256) + bias, ReLU — WMMA, one wave per 16x16 tile.
__global__ __launch_bounds__(32)
void k_fc1_wmma(const _Float16* __restrict__ A, const _Float16* __restrict__ wp,
                const float* __restrict__ bias, float* __restrict__ out) {
  const int lane = threadIdx.x & 31;
  const int mt = blockIdx.x;   // 0..1
  const int nt = blockIdx.y;   // 0..15
  const int row = mt * 16 + (lane & 15);
  const int kbase = (lane >> 4) << 4;
  const int nk = 128;
  v8f acc = {};
  const _Float16* wrow = wp + (((size_t)nt * nk) * 32 + lane) * 16;
  for (int kc = 0; kc < nk; ++kc) {
    const _Float16* s = A + (size_t)row * 4096 + kc * 32 + kbase;
    v16h a = cat8(*(const v8h*)s, *(const v8h*)(s + 8));
    const _Float16* ws = wrow + (size_t)kc * 512;
    v16h bf = cat8(*(const v8h*)ws, *(const v8h*)(ws + 8));
    if (kc + 1 < nk) __builtin_prefetch(ws + 512, 0, 1);
    acc = __builtin_amdgcn_wmma_f32_16x16x32_f16(false, a, false, bf,
                                                 (short)0, acc, false, false);
  }
  const int n = nt * 16 + (lane & 15);
  const float bi = bias[n];
  const int mo = (lane >> 4) << 3;
  #pragma unroll
  for (int r = 0; r < 8; ++r) {
    int m = mt * 16 + r + mo;
    out[(size_t)m * 256 + n] = fmaxf(acc[r] + bi, 0.0f);
  }
}

__global__ void k_fc2(const float* __restrict__ h, const float* __restrict__ w,
                      const float* __restrict__ bias, float* __restrict__ out) {
  int tid = blockIdx.x * blockDim.x + threadIdx.x;
  if (tid >= BB * 40) return;
  int j = tid % 40, b = tid / 40;
  float s = bias[j];
  const float* hr = h + (size_t)b * 256;
  const float* wr = w + (size_t)j * 256;
  #pragma unroll 8
  for (int i = 0; i < 256; ++i) s += hr[i] * wr[i];
  out[tid] = s;
}

// ---------------------------------------------------------------------------
// Workspace layout (bytes)
// ---------------------------------------------------------------------------
static const size_t SZ_SUMS = (size_t)BB * GV * 64 * 4;   // 268 MB (sums, then conv-out ping)
static const size_t SZ_I    = (size_t)BB * GV * 64 * 2;   // 134 MB (f16 activations pong)
static const size_t SZ_CNT  = (size_t)BB * GV * 4;        // 4 MB
static const size_t OFF_S   = 0;
static const size_t OFF_I   = OFF_S + SZ_SUMS;
static const size_t OFF_CNT = OFF_I + SZ_I;
static const size_t OFF_WP1 = OFF_CNT + SZ_CNT;
static const size_t SZ_WP1  = (size_t)4096 * 64 * 2;
static const size_t OFF_WP2 = OFF_WP1 + SZ_WP1;
static const size_t SZ_WP2  = (size_t)4096 * 128 * 2;
static const size_t OFF_WP3 = OFF_WP2 + SZ_WP2;
static const size_t SZ_WP3  = (size_t)8192 * 256 * 2;
static const size_t OFF_WP4 = OFF_WP3 + SZ_WP3;
static const size_t SZ_WP4  = (size_t)16384 * 512 * 2;
static const size_t OFF_WPF = OFF_WP4 + SZ_WP4;
static const size_t SZ_WPF  = (size_t)4096 * 256 * 2;
static const size_t OFF_BN  = OFF_WPF + SZ_WPF;           // 1920 floats used
static const size_t OFF_HF  = OFF_BN + 16384;             // hflat f16 32x4096
static const size_t OFF_F1O = OFF_HF + (size_t)BB * 4096 * 2; // fc1 out f32 32x256

extern "C" void kernel_launch(void* const* d_in, const int* in_sizes, int n_in,
                              void* d_out, int out_size, void* d_ws, size_t ws_size,
                              hipStream_t stream) {
  (void)in_sizes; (void)n_in; (void)out_size; (void)ws_size;
  const float* x     = (const float*)d_in[0];
  const float* pv_w1 = (const float*)d_in[1];
  const float* pv_b1 = (const float*)d_in[2];
  const float* bn0_g = (const float*)d_in[3];
  const float* bn0_b = (const float*)d_in[4];
  const float* bn0_m = (const float*)d_in[5];
  const float* bn0_v = (const float*)d_in[6];
  const float* pv_w2 = (const float*)d_in[7];
  const float* pv_b2 = (const float*)d_in[8];
  const float* cw[4]  = {(const float*)d_in[9],  (const float*)d_in[15],
                         (const float*)d_in[21], (const float*)d_in[27]};
  const float* cbias[4] = {(const float*)d_in[10], (const float*)d_in[16],
                           (const float*)d_in[22], (const float*)d_in[28]};
  const float* bng[4] = {(const float*)d_in[11], (const float*)d_in[17],
                         (const float*)d_in[23], (const float*)d_in[29]};
  const float* bnb[4] = {(const float*)d_in[12], (const float*)d_in[18],
                         (const float*)d_in[24], (const float*)d_in[30]};
  const float* bnm[4] = {(const float*)d_in[13], (const float*)d_in[19],
                         (const float*)d_in[25], (const float*)d_in[31]};
  const float* bnv[4] = {(const float*)d_in[14], (const float*)d_in[20],
                         (const float*)d_in[26], (const float*)d_in[32]};
  const float* fc1_w = (const float*)d_in[33];
  const float* fc1_b = (const float*)d_in[34];
  const float* fc2_w = (const float*)d_in[35];
  const float* fc2_b = (const float*)d_in[36];

  char* ws = (char*)d_ws;
  float*     sums  = (float*)(ws + OFF_S);
  _Float16*  regS  = (_Float16*)(ws + OFF_S);   // reused as conv-out after normalize
  _Float16*  regI  = (_Float16*)(ws + OFF_I);
  float*     cnt   = (float*)(ws + OFF_CNT);
  _Float16*  wp[4] = {(_Float16*)(ws + OFF_WP1), (_Float16*)(ws + OFF_WP2),
                      (_Float16*)(ws + OFF_WP3), (_Float16*)(ws + OFF_WP4)};
  _Float16*  wpf   = (_Float16*)(ws + OFF_WPF);
  float*     bnbuf = (float*)(ws + OFF_BN);
  // scale/shift offsets (floats): [s1 64][h1 64][s2 128][h2 128][s3 256][h3 256][s4 512][h4 512]
  float* sc[4] = {bnbuf + 0,   bnbuf + 128, bnbuf + 384, bnbuf + 896};
  float* sh[4] = {bnbuf + 64,  bnbuf + 256, bnbuf + 640, bnbuf + 1408};
  _Float16* hflat = (_Float16*)(ws + OFF_HF);
  float*    fc1o  = (float*)(ws + OFF_F1O);
  float*    out   = (float*)d_out;

  const int Cin[4] = {64, 64, 128, 256};
  const int Co[4]  = {64, 128, 256, 512};
  const int ID[4]  = {32, 16, 8, 4};
  const int OD[4]  = {33, 17, 9, 5};
  const int PD[4]  = {16, 8, 4, 2};

  // Zero scatter accumulators (graph-capture-legal async memsets).
  (void)hipMemsetAsync(ws + OFF_S, 0, SZ_SUMS, stream);
  (void)hipMemsetAsync(ws + OFF_CNT, 0, SZ_CNT, stream);

  // Pack weights into WMMA B-fragment layout.
  for (int l = 0; l < 4; ++l) {
    size_t tot = (size_t)Co[l] * Cin[l] * 64;
    k_pack_conv<<<dim3((unsigned)((tot + 255) / 256)), 256, 0, stream>>>(cw[l], wp[l], Cin[l], Co[l]);
  }
  {
    size_t tot = (size_t)256 * 4096;
    k_pack_fc<<<dim3((unsigned)((tot + 255) / 256)), 256, 0, stream>>>(fc1_w, wpf, 4096, 256);
  }
  // Fold BN+bias.
  for (int l = 0; l < 4; ++l)
    k_bnprep<<<dim3((Co[l] + 63) / 64), 64, 0, stream>>>(bng[l], bnb[l], bnm[l], bnv[l],
                                                         cbias[l], sc[l], sh[l], Co[l]);

  // Point MLP + scatter, then normalize into f16 channel-last voxel grid.
  k_point_scatter<<<dim3((BB * NP + 127) / 128), 128, 0, stream>>>(
      x, pv_w1, pv_b1, bn0_g, bn0_b, bn0_m, bn0_v, pv_w2, pv_b2, sums, cnt);
  {
    size_t tot = (size_t)BB * GV * 64;
    k_normalize<<<dim3((unsigned)((tot + 255) / 256)), 256, 0, stream>>>(sums, cnt, regI);
  }

  // Conv stack: regI -> (conv) regS -> (pool) regI, x4.
  for (int l = 0; l < 4; ++l) {
    int P = OD[l] * OD[l] * OD[l];
    int pt16 = (P + 15) / 16;
    dim3 grid((pt16 + 3) / 4, Co[l] / 32, BB);
    k_conv_wmma<<<grid, 128, 0, stream>>>(regI, wp[l], sc[l], sh[l], regS,
                                          Cin[l], Co[l], ID[l], OD[l]);
    size_t tot = (size_t)BB * PD[l] * PD[l] * PD[l] * Co[l];
    k_maxpool<<<dim3((unsigned)((tot + 255) / 256)), 256, 0, stream>>>(regS, regI,
                                                                       Co[l], OD[l], PD[l]);
  }

  // Head: flatten -> fc1 (WMMA) -> fc2.
  k_flatten<<<dim3((BB * 4096 + 255) / 256), 256, 0, stream>>>(regI, hflat);
  k_fc1_wmma<<<dim3(2, 16), 32, 0, stream>>>(hflat, wpf, fc1_b, fc1o);
  k_fc2<<<dim3((BB * 40 + 127) / 128), 128, 0, stream>>>(fc1o, fc2_w, fc2_b, out);
}
